// RGCN_DistMult_22857815949324
// MI455X (gfx1250) — matrix operand order repaired and checked by reading, not compiled, not run on hardware
//
#include <hip/hip_runtime.h>
#include <cstdint>

// ---------------------------------------------------------------------------
// RGCN (2 layers, H=128, R=30) for gfx1250: edge-grouped bf16 WMMA GEMMs,
// TDM (tensor_load_to_lds) staging of weight tiles, branch-free epilogues.
// ---------------------------------------------------------------------------

typedef __bf16 bhalf;
typedef __attribute__((ext_vector_type(16))) __bf16 v16bf;
typedef __attribute__((ext_vector_type(8)))  float  v8f;
typedef __attribute__((ext_vector_type(4)))  unsigned int u32x4;
typedef __attribute__((ext_vector_type(8)))  int i32x8;
typedef __attribute__((ext_vector_type(4)))  int i32x4;

union Frag16 { v16bf v; uint4 q[2]; };

constexpr int kH     = 128;   // hidden dim (fixed by problem)
constexpr int kPitch = 152;   // LDS row pitch (bf16): 304B = 76 dwords; 16B-aligned rows,
                              // bank stride 12 -> conflict-free 16-lane b128 reads

static __device__ __forceinline__ bhalf f2bf(float f) {
  unsigned u = __float_as_uint(f);
  unsigned r = u + 0x7FFFu + ((u >> 16) & 1u);   // round-to-nearest-even
  unsigned short s = (unsigned short)(r >> 16);
  return __builtin_bit_cast(bhalf, s);
}

// ---------------------------------------------------------------------------
// TDM: DMA one 128x128 bf16 tile (row-major, stride 128) into LDS with the
// pitch-152 padding applied by the TDM itself (pad every 64 dwords by 12).
// Issue from ONE wave; that wave waits TENSORcnt==0 before the block barrier.
// This toolchain's builtin is the 6-arg clang-23 form:
//   (uint32x4 g0, int32x8 g1, int32x4 g2, int32x4 g3, int32x8 g4, i32 cpol)
// ---------------------------------------------------------------------------
static __device__ __forceinline__ void tdm_load_tile_128x128_bf16(const bhalf* gsrc,
                                                                  unsigned ldsByteAddr) {
#if __has_builtin(__builtin_amdgcn_tensor_load_to_lds)
  unsigned long long ga = (unsigned long long)(uintptr_t)gsrc;
  u32x4 g0;
  g0[0] = 1u;                                               // count=1, user descriptor
  g0[1] = ldsByteAddr;                                      // lds_addr (bytes)
  g0[2] = (unsigned)(ga & 0xFFFFFFFFu);                     // global_addr[31:0]
  g0[3] = (unsigned)((ga >> 32) & 0x1FFFFFFu) | (2u << 30); // addr[56:32] | type=2
  i32x8 g1;
  g1[0] = (1 << 16)     // data_size = 2 bytes
        | (1 << 20)     // pad_enable
        | (5 << 22)     // pad_interval: every 64 dwords (one 256B row)
        | (11 << 25);   // pad_amount: 12 dwords (48B) -> pitch 152 elems
  g1[1] = (kH & 0xFFFF) << 16;     // tensor_dim0[15:0] at bit 48
  g1[2] = (kH << 16);              // tensor_dim0 hi=0 | tensor_dim1[15:0]
  g1[3] = (kH << 16);              // tensor_dim1 hi=0 | tile_dim0=128
  g1[4] = kH;                      // tile_dim1=128, tile_dim2=0
  g1[5] = kH;                      // tensor_dim0_stride = 128 (lo)
  g1[6] = (int)((unsigned)(kH * kH) << 16);  // stride0 hi=0 | stride1 lo
  g1[7] = 0;
  i32x4 gz4 = {0, 0, 0, 0};
  i32x8 gz8 = {0, 0, 0, 0, 0, 0, 0, 0};
  __builtin_amdgcn_tensor_load_to_lds(g0, g1, gz4, gz4, gz8, 0);
#else
  (void)gsrc; (void)ldsByteAddr;
#endif
}

static __device__ __forceinline__ void tdm_wait_all() {
#if __has_builtin(__builtin_amdgcn_s_wait_tensorcnt)
  __builtin_amdgcn_s_wait_tensorcnt(0);
#endif
}

// ---------------------------------------------------------------------------
// small utility kernels
// ---------------------------------------------------------------------------
__global__ void zero_u32_kernel(unsigned* __restrict__ p, long long n) {
  long long i = (long long)blockIdx.x * blockDim.x + threadIdx.x;
  if (i < n) p[i] = 0u;
}

// per-(dst,rel) degree + per-relation histogram
__global__ void count_kernel(const int* __restrict__ dst, const int* __restrict__ et,
                             unsigned* __restrict__ deg, unsigned* __restrict__ relCount,
                             int E, int R) {
  int e = blockIdx.x * blockDim.x + threadIdx.x;
  if (e >= E) return;
  int t = et[e];
  atomicAdd(&deg[(size_t)dst[e] * R + t], 1u);
  atomicAdd(&relCount[t], 1u);
}

// tiny serial scan over R (=30) relations: edge offsets, 64-edge tile offsets, cursors
__global__ void scan_kernel(const unsigned* __restrict__ relCount,
                            int* __restrict__ relOff, int* __restrict__ tileOff,
                            int* __restrict__ cursor, int R) {
  if (threadIdx.x == 0 && blockIdx.x == 0) {
    int acc = 0, tacc = 0;
    for (int r = 0; r < R; ++r) {
      relOff[r] = acc; tileOff[r] = tacc; cursor[r] = acc;
      int c = (int)relCount[r];
      acc += c; tacc += (c + 63) / 64;
    }
    relOff[R] = acc; tileOff[R] = tacc;
  }
}

// counting-sort edge ids by relation
__global__ void scatter_kernel(const int* __restrict__ et, int* __restrict__ cursor,
                               int* __restrict__ sortedEid, int E) {
  int e = blockIdx.x * blockDim.x + threadIdx.x;
  if (e >= E) return;
  int pos = atomicAdd(&cursor[et[e]], 1);
  sortedEid[pos] = e;
}

// h = bf16(emb[x])
__global__ void gather_emb_kernel(const int* __restrict__ x, const float* __restrict__ emb,
                                  bhalf* __restrict__ hbf, int N) {
  int idx = blockIdx.x * blockDim.x + threadIdx.x;
  if (idx >= N * kH) return;
  int i = idx >> 7, c = idx & (kH - 1);
  hbf[idx] = f2bf(emb[(size_t)x[i] * kH + c]);
}

// W[m][k][j] (f32, row-major) -> Wt[m][j][k] (bf16) so B-fragments read contiguous K runs
__global__ void convT_kernel(const float* __restrict__ src, bhalf* __restrict__ dst, int nMat) {
  int idx = blockIdx.x * blockDim.x + threadIdx.x;
  if (idx >= nMat * kH * kH) return;
  int m = idx >> 14, rem = idx & (kH * kH - 1);
  int k = rem >> 7, j = rem & (kH - 1);
  dst[(size_t)m * kH * kH + (size_t)j * kH + k] = f2bf(src[idx]);
}

// ---------------------------------------------------------------------------
// dense root GEMM: out[n, :] = h[n, :] @ root + bias   (64 node rows / block)
// ---------------------------------------------------------------------------
__global__ __launch_bounds__(128) void root_gemm_kernel(
    const bhalf* __restrict__ hbf, const bhalf* __restrict__ rootT,
    const float* __restrict__ bias, float* __restrict__ out, int N) {
  __shared__ bhalf Alds[64][kPitch];
  __shared__ bhalf Blds[kH][kPitch];

  const int tid = threadIdx.x;
  const int rbase = blockIdx.x * 64;
  const int cnt = min(64, N - rbase);
  __builtin_prefetch(hbf + (size_t)(rbase + 64) * kH, 0, 1);  // next block's A rows

  if (tid < 32)                                               // wave 0: DMA B via TDM
    tdm_load_tile_128x128_bf16(rootT, (unsigned)(uintptr_t)&Blds[0][0]);

  const uint4 z4 = {0u, 0u, 0u, 0u};
  for (int q = tid; q < 64 * 16; q += 128) {                  // A rows (zero-fill tail)
    int row = q >> 4, seg = q & 15;
    uint4 v = z4;
    if (row < cnt) v = *((const uint4*)(hbf + (size_t)(rbase + row) * kH) + seg);
    *(uint4*)&Alds[row][seg * 8] = v;
  }
  if (tid < 32) tdm_wait_all();                               // wave 0 fences the DMA
  __syncthreads();

  const int wave = tid >> 5, lane = tid & 31;
  const int half = lane >> 4, l15 = lane & 15;

  Frag16 afrag[4];
  const int arow = wave * 16 + l15;
  const int kbase = half * 8;            // ISA 16-bit A layout: K = kbase..+7, 16+kbase..+7
#pragma unroll
  for (int kc = 0; kc < 4; ++kc) {
    afrag[kc].q[0] = *(const uint4*)&Alds[arow][kc * 32 + kbase];
    afrag[kc].q[1] = *(const uint4*)&Alds[arow][kc * 32 + kbase + 16];
  }

#pragma unroll
  for (int ct = 0; ct < 8; ++ct) {
    v8f acc = {};
    const int brow = ct * 16 + l15;      // Wt row == output column
    const int koff = half * 8;           // B layout: K = koff..+7, 16+koff..+7 (matches A)
#pragma unroll
    for (int kc = 0; kc < 4; ++kc) {
      Frag16 bf;
      bf.q[0] = *(const uint4*)&Blds[brow][kc * 32 + koff];
      bf.q[1] = *(const uint4*)&Blds[brow][kc * 32 + koff + 16];
      acc = __builtin_amdgcn_wmma_f32_16x16x32_bf16(false, afrag[kc].v, false, bf.v,
                                                    (short)0, acc, false, false);
    }
    const int col = ct * 16 + l15;
    const int mbase = wave * 16 + half * 8;
    const float bc = bias[col];
    float* orow = out + (size_t)(rbase + mbase) * kH + col;
    if (cnt == 64) {                      // uniform fast path: unguarded stores
#pragma unroll
      for (int v = 0; v < 8; ++v) orow[(size_t)v * kH] = acc[v] + bc;
    } else {
#pragma unroll
      for (int v = 0; v < 8; ++v)
        if (mbase + v < cnt) orow[(size_t)v * kH] = acc[v] + bc;
    }
  }
}

// ---------------------------------------------------------------------------
// edge GEMM: 64 same-relation sorted edges / block.
// Y = gather(h,src) @ W[r]; out[dst] += Y * (1/deg(dst,r))  via f32 atomics.
// Pad rows: A zero-filled, norm=0, dst clamped to 0 -> branch-free atomics.
// ---------------------------------------------------------------------------
__global__ __launch_bounds__(128) void edge_gemm_kernel(
    const int* __restrict__ sortedEid, const int* __restrict__ src,
    const int* __restrict__ dst, const unsigned* __restrict__ deg,
    const int* __restrict__ relOff, const int* __restrict__ tileOff,
    const bhalf* __restrict__ hbf, const bhalf* __restrict__ WtBf,
    float* __restrict__ out, int R) {
  __shared__ bhalf Alds[64][kPitch];
  __shared__ bhalf Blds[kH][kPitch];
  __shared__ float norm_s[64];
  __shared__ int dst_s[64];
  __shared__ int src_s[64];

  const int b = blockIdx.x;
  if (b >= tileOff[R]) return;
  int r = 0;
  while (tileOff[r + 1] <= b) ++r;              // R=30: cheap scan, L2-resident
  const int k0 = b - tileOff[r];
  const int estart = relOff[r] + 64 * k0;
  const int ecount = min(64, relOff[r + 1] - estart);

  const int tid = threadIdx.x;
  const bhalf* Wr = WtBf + (size_t)r * kH * kH;
  __builtin_prefetch(Wr, 0, 3);                 // global_prefetch_b8

  if (tid < 32)                                 // wave 0: DMA W_r^T tile via TDM
    tdm_load_tile_128x128_bf16(Wr, (unsigned)(uintptr_t)&Blds[0][0]);

  if (tid < 64) {
    int i = tid;
    if (i < ecount) {
      int e = sortedEid[estart + i];
      int s = src[e], d = dst[e];
      src_s[i] = s; dst_s[i] = d;
      unsigned dg = deg[(size_t)d * R + r];
      norm_s[i] = (dg == 0u) ? 1.0f : 1.0f / (float)dg;
    } else { src_s[i] = 0; dst_s[i] = 0; norm_s[i] = 0.0f; }  // pad: adds 0 to row 0
  }
  __syncthreads();

  const uint4 z4 = {0u, 0u, 0u, 0u};
  for (int q = tid; q < 64 * 16; q += 128) {    // gather A rows; zero-fill pad rows
    int row = q >> 4, seg = q & 15;
    uint4 v = z4;
    if (row < ecount) v = *((const uint4*)(hbf + (size_t)src_s[row] * kH) + seg);
    *(uint4*)&Alds[row][seg * 8] = v;
  }
  if (tid < 32) tdm_wait_all();
  __syncthreads();

  const int wave = tid >> 5, lane = tid & 31;
  const int half = lane >> 4, l15 = lane & 15;

  Frag16 afrag[4];
  const int arow = wave * 16 + l15;
  const int kbase = half * 8;
#pragma unroll
  for (int kc = 0; kc < 4; ++kc) {
    afrag[kc].q[0] = *(const uint4*)&Alds[arow][kc * 32 + kbase];
    afrag[kc].q[1] = *(const uint4*)&Alds[arow][kc * 32 + kbase + 16];
  }

#pragma unroll
  for (int ct = 0; ct < 8; ++ct) {
    v8f acc = {};
    const int brow = ct * 16 + l15;
    const int koff = half * 8;
#pragma unroll
    for (int kc = 0; kc < 4; ++kc) {
      Frag16 bf;
      bf.q[0] = *(const uint4*)&Blds[brow][kc * 32 + koff];
      bf.q[1] = *(const uint4*)&Blds[brow][kc * 32 + koff + 16];
      acc = __builtin_amdgcn_wmma_f32_16x16x32_bf16(false, afrag[kc].v, false, bf.v,
                                                    (short)0, acc, false, false);
    }
    const int col = ct * 16 + l15;
    const int mbase = wave * 16 + half * 8;
#pragma unroll
    for (int v = 0; v < 8; ++v) {               // branch-free scaled atomic scatter
      int m = mbase + v;
      atomicAdd(&out[(size_t)dst_s[m] * kH + col], acc[v] * norm_s[m]);
    }
  }
}

// ---------------------------------------------------------------------------
// BatchNorm (training stats, biased var) + ReLU
// ---------------------------------------------------------------------------
__global__ __launch_bounds__(128) void bn_stats_kernel(const float* __restrict__ out,
                                                       float* __restrict__ bnSum,
                                                       float* __restrict__ bnSq, int N) {
  int c = threadIdx.x;                  // one column per thread (H == blockDim.x == 128)
  float s = 0.f, ss = 0.f;
  for (int i = blockIdx.x; i < N; i += gridDim.x) {
    float v = out[(size_t)i * kH + c];
    s += v; ss += v * v;
  }
  atomicAdd(&bnSum[c], s);
  atomicAdd(&bnSq[c], ss);
}

// mode 0: write bf16 h (feeds next layer's WMMA). mode 1: write f32 d_out.
__global__ void bn_apply_kernel(const float* __restrict__ out, const float* __restrict__ bnSum,
                                const float* __restrict__ bnSq, const float* __restrict__ g,
                                const float* __restrict__ be, int N, int mode,
                                bhalf* __restrict__ hbf, float* __restrict__ fout) {
  int idx = blockIdx.x * blockDim.x + threadIdx.x;
  if (idx >= N * kH) return;
  int c = idx & (kH - 1);
  float invN = 1.0f / (float)N;
  float mu = bnSum[c] * invN;
  float var = bnSq[c] * invN - mu * mu;
  float v = (out[idx] - mu) * rsqrtf(var + 1e-5f) * g[c] + be[c];
  v = fmaxf(v, 0.0f);
  if (mode == 0) hbf[idx] = f2bf(v);
  else           fout[idx] = v;
}

// ---------------------------------------------------------------------------
// host orchestration
// ---------------------------------------------------------------------------
extern "C" void kernel_launch(void* const* d_in, const int* in_sizes, int n_in,
                              void* d_out, int out_size, void* d_ws, size_t ws_size,
                              hipStream_t stream) {
  (void)n_in; (void)out_size; (void)ws_size;
  const int N = in_sizes[0];
  const int E = in_sizes[2];
  const int R = in_sizes[4] / (kH * kH);

  const int*   x     = (const int*)d_in[0];
  const int*   ei    = (const int*)d_in[1];
  const int*   et    = (const int*)d_in[2];
  const float* emb   = (const float*)d_in[3];
  const float* Wmat[2]  = {(const float*)d_in[4],  (const float*)d_in[9]};
  const float* rootm[2] = {(const float*)d_in[5],  (const float*)d_in[10]};
  const float* bvec[2]  = {(const float*)d_in[6],  (const float*)d_in[11]};
  const float* gvec[2]  = {(const float*)d_in[7],  (const float*)d_in[12]};
  const float* bevec[2] = {(const float*)d_in[8],  (const float*)d_in[13]};
  const int* srcIdx = ei;
  const int* dstIdx = ei + E;

  // --- workspace carve-out (~93 MB) ---
  char* base = (char*)d_ws;
  size_t off = 0;
  auto take = [&](size_t bytes) -> char* {
    off = (off + 255) & ~(size_t)255;
    char* p = base + off; off += bytes; return p;
  };
  bhalf*    hbf       = (bhalf*)take((size_t)N * kH * sizeof(bhalf));
  float*    outb      = (float*)take((size_t)N * kH * sizeof(float));
  bhalf*    Wtbf      = (bhalf*)take((size_t)R * kH * kH * sizeof(bhalf));
  bhalf*    rootTbf   = (bhalf*)take((size_t)kH * kH * sizeof(bhalf));
  unsigned* deg       = (unsigned*)take((size_t)N * R * sizeof(unsigned));
  unsigned* relCount  = (unsigned*)take((R + 1) * sizeof(unsigned));
  int*      relOff    = (int*)take((R + 1) * sizeof(int));
  int*      tileOff   = (int*)take((R + 1) * sizeof(int));
  int*      cursor    = (int*)take((R + 1) * sizeof(int));
  int*      sortedEid = (int*)take((size_t)E * sizeof(int));
  float*    bnSum     = (float*)take(2 * kH * sizeof(float));
  float*    bnSq      = bnSum + kH;

  auto cdiv = [](long long a, long long b) -> int { return (int)((a + b - 1) / b); };

  // --- graph preprocessing (layer-invariant) ---
  zero_u32_kernel<<<cdiv((long long)N * R, 256), 256, 0, stream>>>(deg, (long long)N * R);
  zero_u32_kernel<<<1, 64, 0, stream>>>(relCount, R);
  count_kernel<<<cdiv(E, 256), 256, 0, stream>>>(dstIdx, et, deg, relCount, E, R);
  scan_kernel<<<1, 32, 0, stream>>>(relCount, relOff, tileOff, cursor, R);
  scatter_kernel<<<cdiv(E, 256), 256, 0, stream>>>(et, cursor, sortedEid, E);
  gather_emb_kernel<<<cdiv((long long)N * kH, 256), 256, 0, stream>>>(x, emb, hbf, N);

  const int tileGrid = (E + 63) / 64 + R;   // upper bound on per-relation 64-edge tiles

  for (int L = 0; L < 2; ++L) {
    convT_kernel<<<cdiv((long long)R * kH * kH, 256), 256, 0, stream>>>(Wmat[L], Wtbf, R);
    convT_kernel<<<cdiv((long long)kH * kH, 256), 256, 0, stream>>>(rootm[L], rootTbf, 1);

    root_gemm_kernel<<<cdiv(N, 64), 128, 0, stream>>>(hbf, rootTbf, bvec[L], outb, N);
    edge_gemm_kernel<<<tileGrid, 128, 0, stream>>>(sortedEid, srcIdx, dstIdx, deg,
                                                   relOff, tileOff, hbf, Wtbf, outb, R);

    zero_u32_kernel<<<1, 256, 0, stream>>>((unsigned*)bnSum, 2 * kH);
    bn_stats_kernel<<<240, 128, 0, stream>>>(outb, bnSum, bnSq, N);
    bn_apply_kernel<<<cdiv((long long)N * kH, 256), 256, 0, stream>>>(
        outb, bnSum, bnSq, gvec[L], bevec[L], N, /*mode=*/L, hbf, (float*)d_out);
  }
}
